// GL_DAC_Model_51135880626481
// MI455X (gfx1250) — matrix operand order repaired and checked by reading, not compiled, not run on hardware
//
#include <hip/hip_runtime.h>

// ---------------- types for WMMA ----------------
typedef __attribute__((ext_vector_type(16))) __bf16 bf16x16;
typedef __attribute__((ext_vector_type(8)))  float  v8f;
typedef __attribute__((ext_vector_type(8)))  float  floatx8;

// ---------------- problem constants ----------------
constexpr int BSZ   = 32;     // batch
constexpr int NP    = 196;    // patches
constexpr int DIM   = 512;    // feature dim
constexpr int NCLS  = 100;    // classes
constexpr int NDV   = 51;     // text descriptors per class
constexpr int KTOP  = 50;     // top-k
constexpr int NCACHE= 800;    // cache images
constexpr int NH    = 8;      // heads
constexpr int DH    = 64;     // head dim
constexpr int DFF   = 2048;   // mlp hidden
constexpr int CHUNK = 100;    // cache images per chunk
constexpr float LAM  = 0.5f;
constexpr float BETA = 5.5f;
constexpr float EPSV = 1e-5f;

// output section offsets (floats)
constexpr size_t OFF_LOGITS = 0;
constexpr size_t OFF_INTER  = 3200;
constexpr size_t OFF_INTRA  = 6400;
constexpr size_t OFF_AG     = 9600;
constexpr size_t OFF_TOPK   = 25984;
constexpr size_t OFF_BOTK   = 845184;
constexpr size_t OFF_RN     = 1664384;

// ---------------- device helpers ----------------
__device__ __forceinline__ float blk_sum(float v, float* red) {
    int t = threadIdx.x;
    red[t] = v; __syncthreads();
    for (int s = blockDim.x >> 1; s > 0; s >>= 1) {
        if (t < s) red[t] += red[t + s];
        __syncthreads();
    }
    float r = red[0]; __syncthreads();
    return r;
}

// block-wide argmax over vals[0..count) with mask; jax tie-break (lowest index)
__device__ __forceinline__ int blk_argmax(const float* vals, const int* taken, int count,
                                          float* rv, int* ri) {
    int t = threadIdx.x;
    float bv = -3.4e38f; int bi = 0x7fffffff;
    for (int p = t; p < count; p += blockDim.x) {
        if (!taken[p]) {
            float v = vals[p];
            if (v > bv || (v == bv && p < bi)) { bv = v; bi = p; }
        }
    }
    rv[t] = bv; ri[t] = bi; __syncthreads();
    for (int s = blockDim.x >> 1; s > 0; s >>= 1) {
        if (t < s) {
            float ov = rv[t + s]; int oi = ri[t + s];
            if (ov > rv[t] || (ov == rv[t] && oi < ri[t])) { rv[t] = ov; ri[t] = oi; }
        }
        __syncthreads();
    }
    int r = ri[0]; __syncthreads();
    return r;
}

__device__ __forceinline__ float gelu_t(float x) {  // jax default tanh approx
    float x3 = x * x * x;
    return 0.5f * x * (1.f + tanhf(0.7978845608f * (x + 0.044715f * x3)));
}

// -------- single WMMA 16x16 tile (clamped loads; caller guards stores) --------
// A fragment: lane<16 row=l, K 0..15 contiguous; lane>=16 row=l-16, K 16..31 contiguous
// B fragment: lane<16 col=l, K 0..15 contiguous; lane>=16 col=l-16, K 16..31 contiguous
__device__ __forceinline__ v8f wmma_tile(const float* __restrict__ A, int lda, int M,
                                         const __bf16* __restrict__ Bc, int ldb, int N,
                                         int K, int tm, int tn) {
    int lane = threadIdx.x & 31;
    int hlf  = lane >> 4;
    int l15  = lane & 15;
    int arow = tm * 16 + l15; if (arow >= M) arow = M - 1;   // clamp: padded output
    int bcol = tn * 16 + l15; if (bcol >= N) bcol = N - 1;   // rows/cols never stored
    const float*  ap = A  + (size_t)arow * lda + hlf * 16;
    const __bf16* bp = Bc + (size_t)bcol * ldb + hlf * 16;
    v8f acc = (v8f)0.0f;
    for (int kk = 0; kk < K; kk += 32) {
        floatx8 f0 = *(const floatx8*)(ap + kk);
        floatx8 f1 = *(const floatx8*)(ap + kk + 8);
        bf16x16 a;
#pragma unroll
        for (int i = 0; i < 8; i++) { a[i] = (__bf16)f0[i]; a[8 + i] = (__bf16)f1[i]; }
        bf16x16 b = *(const bf16x16*)(bp + kk);
        acc = __builtin_amdgcn_wmma_f32_16x16x32_bf16(false, a, false, b,
                                                      (short)0, acc, false, false);
    }
    return acc;
}

// -------- generic batched GEMM: one wave computes a 16x64 strip (4 tiles) --------
// C = A @ B (+bias)(+addend)(gelu); flags: 1=bias[n], 2=gelu, 4=addend (ld = ldc)
// grid: (ceil(M/16), ceil(N/64), batch), block 32
__global__ void k_gemm(const float* __restrict__ A, const __bf16* __restrict__ Bc,
                       float* __restrict__ C, const float* __restrict__ bias,
                       const float* __restrict__ addend,
                       int M, int N, int K, int lda, int ldb, int ldc,
                       long long sA, long long sC, long long sAdd, int flags) {
    int z = blockIdx.z;
    const float* Az = A + (long long)z * sA;
    float* Cz = C + (long long)z * sC;
    const float* addz = (flags & 4) ? addend + (long long)z * sAdd : nullptr;
    int tm = blockIdx.x;
    int lane = threadIdx.x & 31;
    int hlf  = lane >> 4;
    int l15  = lane & 15;
    int arow = tm * 16 + l15; if (arow >= M) arow = M - 1;
    const float* ap = Az + (size_t)arow * lda + hlf * 16;
    const __bf16* bp[4];
    int cols[4];
#pragma unroll
    for (int j = 0; j < 4; j++) {
        int col = blockIdx.y * 64 + j * 16 + l15;
        cols[j] = col;
        int cc = col < N ? col : N - 1;
        bp[j] = Bc + (size_t)cc * ldb + hlf * 16;
    }
    v8f acc[4];
#pragma unroll
    for (int j = 0; j < 4; j++) acc[j] = (v8f)0.0f;
    for (int kk = 0; kk < K; kk += 32) {
        floatx8 f0 = *(const floatx8*)(ap + kk);
        floatx8 f1 = *(const floatx8*)(ap + kk + 8);
        if (kk + 32 < K) __builtin_prefetch(ap + kk + 32, 0, 3);
        bf16x16 a;
#pragma unroll
        for (int i = 0; i < 8; i++) { a[i] = (__bf16)f0[i]; a[8 + i] = (__bf16)f1[i]; }
#pragma unroll
        for (int j = 0; j < 4; j++) {   // 4 independent WMMAs hide the D->C hazard
            bf16x16 b = *(const bf16x16*)(bp[j] + kk);
            acc[j] = __builtin_amdgcn_wmma_f32_16x16x32_bf16(false, a, false, b,
                                                             (short)0, acc[j], false, false);
        }
    }
    int mb = hlf * 8;
#pragma unroll
    for (int j = 0; j < 4; j++) {
        int col = cols[j];
        if (col < N) {
#pragma unroll
            for (int i = 0; i < 8; i++) {
                int m = tm * 16 + mb + i;
                if (m < M) {
                    float x = acc[j][i];
                    if (flags & 1) x += bias[col];
                    if (flags & 4) x += addz[(size_t)m * ldc + col];
                    if (flags & 2) x = gelu_t(x);
                    Cz[(size_t)m * ldc + col] = x;
                }
            }
        }
    }
}

// -------- conversions --------
__global__ void k_cvt(const float* __restrict__ s, __bf16* __restrict__ d, int n) {
    int i = blockIdx.x * blockDim.x + threadIdx.x;
    if (i < n) d[i] = (__bf16)s[i];
}
__global__ void k_cvt_t(const float* __restrict__ s, __bf16* __restrict__ d, int Kd, int Nd) {
    int i = blockIdx.x * blockDim.x + threadIdx.x;
    if (i < Kd * Nd) { int k = i / Nd, n = i % Nd; d[(size_t)n * Kd + k] = (__bf16)s[i]; }
}

// -------- base_text_mean: norm(mean over ND of text_cache) --------
__global__ void k_btm(const float* __restrict__ tc, float* __restrict__ btm) {
    __shared__ float sb[DIM]; __shared__ float red[256];
    int c = blockIdx.x, t = threadIdx.x;
    float ssq = 0.f;
    for (int d = t; d < DIM; d += 256) {
        float s = 0.f;
        for (int nd = 0; nd < NDV; nd++) s += tc[((size_t)c * NDV + nd) * DIM + d];
        s *= (1.0f / NDV);
        sb[d] = s; ssq += s * s;
    }
    float inv = 1.f / sqrtf(blk_sum(ssq, red));
    for (int d = t; d < DIM; d += 256) btm[(size_t)c * DIM + d] = sb[d] * inv;
}

// -------- in-place row L2-normalize --------
__global__ void k_rownorm(float* __restrict__ X, int ld) {
    __shared__ float red[256];
    float* row = X + (size_t)blockIdx.x * ld;
    int t = threadIdx.x;
    float ssq = 0.f;
    for (int d = t; d < ld; d += 256) { float v = row[d]; ssq += v * v; }
    float inv = 1.f / sqrtf(blk_sum(ssq, red));
    for (int d = t; d < ld; d += 256) row[d] *= inv;
}

// -------- adapted_global = norm(mean over P) --------
__global__ void k_ag(const float* __restrict__ ad, float* __restrict__ oag, float* __restrict__ wag) {
    __shared__ float sb[DIM]; __shared__ float red[256];
    int b = blockIdx.x, t = threadIdx.x;
    float ssq = 0.f;
    for (int d = t; d < DIM; d += 256) {
        float s = 0.f;
        for (int p = 0; p < NP; p++) s += ad[((size_t)b * NP + p) * DIM + d];
        s *= (1.0f / NP);
        sb[d] = s; ssq += s * s;
    }
    float inv = 1.f / sqrtf(blk_sum(ssq, red));
    for (int d = t; d < DIM; d += 256) {
        float v = sb[d] * inv;
        oag[(size_t)b * DIM + d] = v; wag[(size_t)b * DIM + d] = v;
    }
}

// -------- sim_target[b,p] = adapted[b,p] . btm[labels[b]] --------
__global__ void k_simt(const float* __restrict__ ad, const float* __restrict__ btm,
                       const int* __restrict__ labels, float* __restrict__ simt) {
    int b = blockIdx.x, t = threadIdx.x;
    int lab = labels[b];
    const float* m = btm + (size_t)lab * DIM;
    for (int p = t; p < NP; p += blockDim.x) {
        const float* r = ad + ((size_t)b * NP + p) * DIM;
        float s = 0.f;
        for (int d = 0; d < DIM; d++) s += r[d] * m[d];
        simt[b * NP + p] = s;
    }
}

// -------- top/bot-k selection, gather feats, query_keys --------
__global__ void k_topk(const float* __restrict__ simt, const float* __restrict__ ad,
                       float* __restrict__ otop, float* __restrict__ obot,
                       float* __restrict__ tkf, float* __restrict__ qk) {
    __shared__ float sv[NP]; __shared__ int taken[NP];
    __shared__ float rv[256]; __shared__ int ri[256];
    __shared__ int idxs[KTOP];
    __shared__ float sb[DIM]; __shared__ float red[256];
    int b = blockIdx.x, t = threadIdx.x;
    for (int p = t; p < NP; p += 256) { sv[p] = simt[b * NP + p]; taken[p] = 0; }
    __syncthreads();
    for (int k = 0; k < KTOP; k++) {
        int bi = blk_argmax(sv, taken, NP, rv, ri);
        if (t == 0) { idxs[k] = bi; taken[bi] = 1; }
        __syncthreads();
    }
    for (int k = 0; k < KTOP; k++) {
        int p = idxs[k];
        const float* src = ad + ((size_t)b * NP + p) * DIM;
        float* d1 = otop + ((size_t)b * KTOP + k) * DIM;
        float* d2 = tkf  + ((size_t)b * KTOP + k) * DIM;
        for (int d = t; d < DIM; d += 256) { float v = src[d]; d1[d] = v; d2[d] = v; }
    }
    __syncthreads();
    float ssq = 0.f;
    for (int d = t; d < DIM; d += 256) {
        float s = 0.f;
        for (int k = 0; k < KTOP; k++) s += tkf[((size_t)b * KTOP + k) * DIM + d];
        s *= (1.0f / KTOP);
        sb[d] = s; ssq += s * s;
    }
    float inv = 1.f / sqrtf(blk_sum(ssq, red));
    for (int d = t; d < DIM; d += 256) qk[(size_t)b * DIM + d] = sb[d] * inv;
    __syncthreads();
    // bottom-k
    for (int p = t; p < NP; p += 256) { sv[p] = -simt[b * NP + p]; taken[p] = 0; }
    __syncthreads();
    for (int k = 0; k < KTOP; k++) {
        int bi = blk_argmax(sv, taken, NP, rv, ri);
        if (t == 0) { idxs[k] = bi; taken[bi] = 1; }
        __syncthreads();
    }
    for (int k = 0; k < KTOP; k++) {
        int p = idxs[k];
        const float* src = ad + ((size_t)b * NP + p) * DIM;
        float* d1 = obot + ((size_t)b * KTOP + k) * DIM;
        for (int d = t; d < DIM; d += 256) d1[d] = src[d];
    }
}

// -------- gather l2p = text_cache[labels[b]] --------
__global__ void k_l2p(const float* __restrict__ tc, const int* __restrict__ labels,
                      float* __restrict__ l2p) {
    int b = blockIdx.x, t = threadIdx.x;
    int c = labels[b];
    const float* src = tc + (size_t)c * NDV * DIM;
    float* dst = l2p + (size_t)b * NDV * DIM;
    for (int i = t; i < NDV * DIM; i += blockDim.x) dst[i] = src[i];
}

// -------- LayerNorm over last dim --------
__global__ void k_ln(const float* __restrict__ x, const float* __restrict__ g,
                     const float* __restrict__ bb, float* __restrict__ y) {
    __shared__ float red[256]; __shared__ float row[DIM];
    int r = blockIdx.x, t = threadIdx.x;
    const float* xr = x + (size_t)r * DIM;
    float s = 0.f, s2 = 0.f;
    for (int d = t; d < DIM; d += 256) { float v = xr[d]; row[d] = v; s += v; s2 += v * v; }
    float mean = blk_sum(s, red) * (1.f / DIM);
    float var  = blk_sum(s2, red) * (1.f / DIM) - mean * mean;
    float inv  = rsqrtf(var + EPSV);
    for (int d = t; d < DIM; d += 256)
        y[(size_t)r * DIM + d] = (row[d] - mean) * inv * g[d] + bb[d];
}

// -------- attention per (b,h): softmax(q k^T / 8) v --------
__global__ void k_attn(const float* __restrict__ q, const float* __restrict__ k,
                       const float* __restrict__ v, float* __restrict__ o) {
    __shared__ float sc[NDV * KTOP];
    int bh = blockIdx.x; int b = bh / NH; int h = bh % NH; int t = threadIdx.x;
    const float* qb = q + (size_t)b * NDV * DIM + h * DH;
    const float* kb = k + (size_t)b * KTOP * DIM + h * DH;
    const float* vb = v + (size_t)b * KTOP * DIM + h * DH;
    for (int i = t; i < NDV * KTOP; i += blockDim.x) {
        int n = i / KTOP, m = i % KTOP;
        float s = 0.f;
        for (int d = 0; d < DH; d++) s += qb[(size_t)n * DIM + d] * kb[(size_t)m * DIM + d];
        sc[i] = s * 0.125f;  // 1/sqrt(64)
    }
    __syncthreads();
    if (t < NDV) {
        float mx = -3.4e38f;
        for (int m = 0; m < KTOP; m++) mx = fmaxf(mx, sc[t * KTOP + m]);
        float sum = 0.f;
        for (int m = 0; m < KTOP; m++) { float e = expf(sc[t * KTOP + m] - mx); sc[t * KTOP + m] = e; sum += e; }
        float inv = 1.f / sum;
        for (int m = 0; m < KTOP; m++) sc[t * KTOP + m] *= inv;
    }
    __syncthreads();
    for (int i = t; i < NDV * DH; i += blockDim.x) {
        int n = i / DH, d = i % DH;
        float s = 0.f;
        for (int m = 0; m < KTOP; m++) s += sc[n * KTOP + m] * vb[(size_t)m * DIM + d];
        o[(size_t)b * NDV * DIM + (size_t)n * DIM + h * DH + d] = s;
    }
}

// -------- refined_norm rows: text_cache + LAM*bias_avg, normalized --------
__global__ void k_refine(const float* __restrict__ tc, const float* __restrict__ tbias,
                         const int* __restrict__ labels, float* __restrict__ rn,
                         __bf16* __restrict__ bfrn, float* __restrict__ orn) {
    __shared__ int slab[BSZ]; __shared__ float row[DIM]; __shared__ float red[256];
    int r = blockIdx.x; int c = r / NDV; int nd = r % NDV; int t = threadIdx.x;
    if (t < BSZ) slab[t] = labels[t];
    __syncthreads();
    float ssq = 0.f;
    for (int d = t; d < DIM; d += 256) {
        float acc = 0.f;
        for (int b = 0; b < BSZ; b++)
            if (slab[b] == c) acc += tbias[((size_t)b * NDV + nd) * DIM + d];
        float v = tc[(size_t)r * DIM + d] + (LAM / BSZ) * acc;
        row[d] = v; ssq += v * v;
    }
    float inv = 1.f / sqrtf(blk_sum(ssq, red));
    for (int d = t; d < DIM; d += 256) {
        float v = row[d] * inv;
        rn[(size_t)r * DIM + d] = v;
        bfrn[(size_t)r * DIM + d] = (__bf16)v;
        orn[(size_t)r * DIM + d] = v;
    }
}

// -------- refined_mean_norm --------
__global__ void k_rmn(const float* __restrict__ rn, float* __restrict__ rmn) {
    __shared__ float sb[DIM]; __shared__ float red[256];
    int c = blockIdx.x, t = threadIdx.x;
    float ssq = 0.f;
    for (int d = t; d < DIM; d += 256) {
        float s = 0.f;
        for (int nd = 0; nd < NDV; nd++) s += rn[((size_t)c * NDV + nd) * DIM + d];
        s *= (1.f / NDV);
        sb[d] = s; ssq += s * s;
    }
    float inv = 1.f / sqrtf(blk_sum(ssq, red));
    for (int d = t; d < DIM; d += 256) rmn[(size_t)c * DIM + d] = sb[d] * inv;
}

// -------- v[n,c] = softmax_n( rmn[c] . rn[c,n] ), stored vmat[c*NDV+n] --------
__global__ void k_vmat(const float* __restrict__ rmn, const float* __restrict__ rn,
                       float* __restrict__ vmat) {
    __shared__ float e[NDV];
    int c = blockIdx.x, t = threadIdx.x;
    if (t < NDV) {
        float s = 0.f;
        const float* a = rmn + (size_t)c * DIM;
        const float* b = rn + ((size_t)c * NDV + t) * DIM;
        for (int d = 0; d < DIM; d++) s += a[d] * b[d];
        e[t] = s;
    }
    __syncthreads();
    if (t == 0) {
        float mx = -3.4e38f;
        for (int n = 0; n < NDV; n++) mx = fmaxf(mx, e[n]);
        float sum = 0.f;
        for (int n = 0; n < NDV; n++) { float x = expf(e[n] - mx); e[n] = x; sum += x; }
        float inv = 1.f / sum;
        for (int n = 0; n < NDV; n++) vmat[c * NDV + n] = e[n] * inv;
    }
}

// -------- base[b,c] = ag[b] . rmn[c] --------
__global__ void k_base(const float* __restrict__ ag, const float* __restrict__ rmn,
                       float* __restrict__ base_) {
    int b = blockIdx.x, t = threadIdx.x;
    if (t < NCLS) {
        float s = 0.f;
        const float* a = ag + (size_t)b * DIM;
        const float* m = rmn + (size_t)t * DIM;
        for (int d = 0; d < DIM; d++) s += a[d] * m[d];
        base_[b * NCLS + t] = s;
    }
}

// -------- w[b,p] = ag[b] . adapted[b,p] --------
__global__ void k_w(const float* __restrict__ ag, const float* __restrict__ ad,
                    float* __restrict__ wb) {
    int b = blockIdx.x, t = threadIdx.x;
    const float* a = ag + (size_t)b * DIM;
    for (int p = t; p < NP; p += blockDim.x) {
        const float* l = ad + ((size_t)b * NP + p) * DIM;
        float s = 0.f;
        for (int d = 0; d < DIM; d++) s += a[d] * l[d];
        wb[b * NP + p] = s;
    }
}

// -------- w_topk: top-50 p of (local . rn[0,0]), gather w, softmax --------
__global__ void k_wtopk(const float* __restrict__ ad, const float* __restrict__ rn0,
                        const float* __restrict__ wb, float* __restrict__ wtopk) {
    __shared__ float sv[NP]; __shared__ int taken[NP];
    __shared__ float rv[256]; __shared__ int ri[256];
    __shared__ float gw[KTOP];
    int b = blockIdx.x, t = threadIdx.x;
    for (int p = t; p < NP; p += 256) {
        const float* l = ad + ((size_t)b * NP + p) * DIM;
        float s = 0.f;
        for (int d = 0; d < DIM; d++) s += l[d] * rn0[d];
        sv[p] = s; taken[p] = 0;
    }
    __syncthreads();
    for (int k = 0; k < KTOP; k++) {
        int bi = blk_argmax(sv, taken, NP, rv, ri);
        if (t == 0) { taken[bi] = 1; gw[k] = wb[b * NP + bi]; }
        __syncthreads();
    }
    if (t == 0) {
        float mx = -3.4e38f;
        for (int k = 0; k < KTOP; k++) mx = fmaxf(mx, gw[k]);
        float sum = 0.f;
        for (int k = 0; k < KTOP; k++) { float e = expf(gw[k] - mx); gw[k] = e; sum += e; }
        float inv = 1.f / sum;
        for (int k = 0; k < KTOP; k++) wtopk[b * KTOP + k] = gw[k] * inv;
    }
}

// -------- fused dense-logits core: per (b,c) WMMA 196x51 scores + per-n top-50 --------
constexpr int SLD = 52;
__global__ void k_dense(const float* __restrict__ ad, const __bf16* __restrict__ bfrn,
                        const float* __restrict__ vmat, const float* __restrict__ wtopk,
                        const float* __restrict__ base_, float* __restrict__ inter_raw) {
    __shared__ float S[NP * SLD];
    __shared__ float tot;
    int b = blockIdx.x, c = blockIdx.y;
    int t = threadIdx.x, w = t >> 5;
    const float* A = ad + (size_t)b * NP * DIM;
    const __bf16* B = bfrn + (size_t)c * NDV * DIM;
    const int MT = (NP + 15) / 16;   // 13
    const int NT = (NDV + 15) / 16;  // 4
    for (int tile = w; tile < MT * NT; tile += 8) {
        int tm = tile / NT, tn = tile % NT;
        v8f acc = wmma_tile(A, DIM, NP, B, DIM, NDV, DIM, tm, tn);
        int lane = t & 31;
        int col  = tn * 16 + (lane & 15);
        int mb   = (lane >> 4) * 8;
        if (col < NDV) {
#pragma unroll
            for (int i = 0; i < 8; i++) {
                int m = tm * 16 + mb + i;
                if (m < NP) S[m * SLD + col] = acc[i];
            }
        }
    }
    if (t == 0) tot = 0.f;
    __syncthreads();
    if (t < NDV) {
        int n = t;
        float acc = 0.f;
        for (int k = 0; k < KTOP; k++) {
            float bv = -3.4e38f; int bi = 0;
            for (int p = 0; p < NP; p++) { float v = S[p * SLD + n]; if (v > bv) { bv = v; bi = p; } }
            acc += wtopk[b * KTOP + k] * bv;
            S[bi * SLD + n] = -3.4e38f;
        }
        atomicAdd(&tot, vmat[c * NDV + n] * acc);
    }
    __syncthreads();
    if (t == 0) inter_raw[b * NCLS + c] = base_[b * NCLS + c] + tot;
}

// -------- cache keys per image (chunked) --------
__global__ void k_ckeys(const float* __restrict__ craw, const float* __restrict__ btm,
                        const int* __restrict__ clab, int ch, float* __restrict__ ckeys) {
    __shared__ float sim[NP]; __shared__ float rinv[NP]; __shared__ int taken[NP];
    __shared__ float rv[256]; __shared__ int ri[256];
    __shared__ float sb[DIM]; __shared__ float red[256];
    __shared__ int idxs[KTOP];
    int i = blockIdx.x; int nc = ch * CHUNK + i; int t = threadIdx.x;
    int lab = clab[nc];
    const float* R = craw + (size_t)i * NP * DIM;
    const float* m = btm + (size_t)lab * DIM;
    for (int p = t; p < NP; p += 256) {
        const float* r = R + (size_t)p * DIM;
        float ssq = 0.f, dot = 0.f;
        for (int d = 0; d < DIM; d++) { float x = r[d]; ssq += x * x; dot += x * m[d]; }
        float iv = 1.f / sqrtf(ssq);
        rinv[p] = iv; sim[p] = dot * iv; taken[p] = 0;
    }
    __syncthreads();
    for (int k = 0; k < KTOP; k++) {
        int bi = blk_argmax(sim, taken, NP, rv, ri);
        if (t == 0) { taken[bi] = 1; idxs[k] = bi; }
        __syncthreads();
    }
    float ssq = 0.f;
    for (int d = t; d < DIM; d += 256) {
        float s = 0.f;
        for (int k = 0; k < KTOP; k++) { int p = idxs[k]; s += R[(size_t)p * DIM + d] * rinv[p]; }
        s *= (1.0f / KTOP);
        sb[d] = s; ssq += s * s;
    }
    float inv = 1.f / sqrtf(blk_sum(ssq, red));
    for (int d = t; d < DIM; d += 256) ckeys[(size_t)nc * DIM + d] = sb[d] * inv;
}

// -------- intra = exp(-BETA*(1 - qk.ck)) @ cache_values --------
__global__ void k_intra(const float* __restrict__ qk, const float* __restrict__ ck,
                        const float* __restrict__ cv, float* __restrict__ intra) {
    __shared__ float aff[NCACHE];
    int b = blockIdx.x, t = threadIdx.x;
    const float* q = qk + (size_t)b * DIM;
    for (int nc = t; nc < NCACHE; nc += 256) {
        const float* kk = ck + (size_t)nc * DIM;
        float s = 0.f;
        for (int d = 0; d < DIM; d++) s += q[d] * kk[d];
        aff[nc] = expf(-BETA * (1.f - s));
    }
    __syncthreads();
    for (int c = t; c < NCLS; c += 256) {
        float s = 0.f;
        for (int nc = 0; nc < NCACHE; nc++) s += aff[nc] * cv[(size_t)nc * NCLS + c];
        intra[b * NCLS + c] = s;
    }
}

// -------- final combine --------
__global__ void k_final(const float* __restrict__ inter_raw, const float* __restrict__ intra,
                        const float* __restrict__ lsp, const float* __restrict__ esp,
                        float* __restrict__ logits, float* __restrict__ inter) {
    int i = blockIdx.x * blockDim.x + threadIdx.x;
    if (i < BSZ * NCLS) {
        float ls = expf(lsp[0]);
        float sc = 1.f / (1.f + expf(-esp[0]));
        float itr = ls * inter_raw[i];
        inter[i] = itr;
        logits[i] = itr * (1.f - sc) + intra[i] * ls * sc;
    }
}

// =====================================================================
extern "C" void kernel_launch(void* const* d_in, const int* in_sizes, int n_in,
                              void* d_out, int out_size, void* d_ws, size_t ws_size,
                              hipStream_t stream) {
    (void)in_sizes; (void)n_in; (void)out_size; (void)ws_size;
    const float* local = (const float*)d_in[0];
    const float* Wa    = (const float*)d_in[1];
    const float* tc    = (const float*)d_in[2];
    const float* Wq    = (const float*)d_in[3];
    const float* Wk    = (const float*)d_in[4];
    const float* Wv    = (const float*)d_in[5];
    const float* Wo    = (const float*)d_in[6];
    const float* ln1g  = (const float*)d_in[7];
    const float* ln1b  = (const float*)d_in[8];
    const float* ln2g  = (const float*)d_in[9];
    const float* ln2b  = (const float*)d_in[10];
    const float* W1    = (const float*)d_in[11];
    const float* b1    = (const float*)d_in[12];
    const float* W2    = (const float*)d_in[13];
    const float* b2    = (const float*)d_in[14];
    const float* lsp   = (const float*)d_in[15];
    const float* esp   = (const float*)d_in[16];
    const float* rawc  = (const float*)d_in[17];
    const float* cv    = (const float*)d_in[18];
    const int*   labels= (const int*)d_in[19];
    const int*   clab  = (const int*)d_in[20];

    float* out = (float*)d_out;
    float* o_logits = out + OFF_LOGITS;
    float* o_inter  = out + OFF_INTER;
    float* o_intra  = out + OFF_INTRA;
    float* o_ag     = out + OFF_AG;
    float* o_top    = out + OFF_TOPK;
    float* o_bot    = out + OFF_BOTK;
    float* o_rn     = out + OFF_RN;

    char* wsb = (char*)d_ws;
    size_t off = 0;
    auto alloc = [&](size_t bytes) -> void* {
        void* p = wsb + off;
        off += (bytes + 255) & ~(size_t)255;
        return p;
    };
    __bf16* bf_Wa = (__bf16*)alloc((size_t)DIM * DIM * 2);
    __bf16* bf_Wq = (__bf16*)alloc((size_t)DIM * DIM * 2);
    __bf16* bf_Wk = (__bf16*)alloc((size_t)DIM * DIM * 2);
    __bf16* bf_Wv = (__bf16*)alloc((size_t)DIM * DIM * 2);
    __bf16* bf_Wo = (__bf16*)alloc((size_t)DIM * DIM * 2);
    __bf16* bf_W1 = (__bf16*)alloc((size_t)DIM * DFF * 2);
    __bf16* bf_W2 = (__bf16*)alloc((size_t)DFF * DIM * 2);
    float* btm    = (float*)alloc((size_t)NCLS * DIM * 4);
    float* adapted= (float*)alloc((size_t)BSZ * NP * DIM * 4);
    float* wag    = (float*)alloc((size_t)BSZ * DIM * 4);
    float* simt   = (float*)alloc((size_t)BSZ * NP * 4);
    float* tkf    = (float*)alloc((size_t)BSZ * KTOP * DIM * 4);
    float* qkbuf  = (float*)alloc((size_t)BSZ * DIM * 4);
    float* l2p    = (float*)alloc((size_t)BSZ * NDV * DIM * 4);
    float* ln1x   = (float*)alloc((size_t)BSZ * NDV * DIM * 4);
    float* qb     = (float*)alloc((size_t)BSZ * NDV * DIM * 4);
    float* kb     = (float*)alloc((size_t)BSZ * KTOP * DIM * 4);
    float* vb     = (float*)alloc((size_t)BSZ * KTOP * DIM * 4);
    float* ob     = (float*)alloc((size_t)BSZ * NDV * DIM * 4);
    float* xres   = (float*)alloc((size_t)BSZ * NDV * DIM * 4);
    float* ln2x   = (float*)alloc((size_t)BSZ * NDV * DIM * 4);
    float* hid    = (float*)alloc((size_t)BSZ * NDV * DFF * 4);
    float* tbias  = (float*)alloc((size_t)BSZ * NDV * DIM * 4);
    float* rn     = (float*)alloc((size_t)NCLS * NDV * DIM * 4);
    __bf16* bf_rn = (__bf16*)alloc((size_t)NCLS * NDV * DIM * 2);
    float* rmn    = (float*)alloc((size_t)NCLS * DIM * 4);
    float* vmat   = (float*)alloc((size_t)NCLS * NDV * 4);
    float* basew  = (float*)alloc((size_t)BSZ * NCLS * 4);
    float* wbuf   = (float*)alloc((size_t)BSZ * NP * 4);
    float* wtopk  = (float*)alloc((size_t)BSZ * KTOP * 4);
    float* interR = (float*)alloc((size_t)BSZ * NCLS * 4);
    float* craw   = (float*)alloc((size_t)CHUNK * NP * DIM * 4);
    float* ckeys  = (float*)alloc((size_t)NCACHE * DIM * 4);

    // --- weight conversions (B stored column-major bf16: Bc[n*K+k] = B[k][n]) ---
    k_cvt<<<(DIM * DIM + 255) / 256, 256, 0, stream>>>(Wa, bf_Wa, DIM * DIM);   // for x @ Wa.T
    k_cvt_t<<<(DIM * DIM + 255) / 256, 256, 0, stream>>>(Wq, bf_Wq, DIM, DIM);
    k_cvt_t<<<(DIM * DIM + 255) / 256, 256, 0, stream>>>(Wk, bf_Wk, DIM, DIM);
    k_cvt_t<<<(DIM * DIM + 255) / 256, 256, 0, stream>>>(Wv, bf_Wv, DIM, DIM);
    k_cvt_t<<<(DIM * DIM + 255) / 256, 256, 0, stream>>>(Wo, bf_Wo, DIM, DIM);
    k_cvt_t<<<(DIM * DFF + 255) / 256, 256, 0, stream>>>(W1, bf_W1, DIM, DFF);
    k_cvt_t<<<(DFF * DIM + 255) / 256, 256, 0, stream>>>(W2, bf_W2, DFF, DIM);
    k_btm<<<NCLS, 256, 0, stream>>>(tc, btm);

    // --- adapted_local = norm(local @ Wa.T) ---
    k_gemm<<<dim3((BSZ * NP) / 16, DIM / 64, 1), 32, 0, stream>>>(
        local, bf_Wa, adapted, nullptr, nullptr,
        BSZ * NP, DIM, DIM, DIM, DIM, DIM, 0, 0, 0, 0);
    k_rownorm<<<BSZ * NP, 256, 0, stream>>>(adapted, DIM);
    k_ag<<<BSZ, 256, 0, stream>>>(adapted, o_ag, wag);
    k_simt<<<BSZ, 256, 0, stream>>>(adapted, btm, labels, simt);
    k_topk<<<BSZ, 256, 0, stream>>>(simt, adapted, o_top, o_bot, tkf, qkbuf);

    // --- bonder ---
    k_l2p<<<BSZ, 256, 0, stream>>>(tc, labels, l2p);
    k_ln<<<BSZ * NDV, 256, 0, stream>>>(l2p, ln1g, ln1b, ln1x);
    k_gemm<<<dim3((NDV + 15) / 16, DIM / 64, BSZ), 32, 0, stream>>>(
        ln1x, bf_Wq, qb, nullptr, nullptr, NDV, DIM, DIM, DIM, DIM, DIM,
        (long long)NDV * DIM, (long long)NDV * DIM, 0, 0);
    k_gemm<<<dim3((KTOP + 15) / 16, DIM / 64, BSZ), 32, 0, stream>>>(
        tkf, bf_Wk, kb, nullptr, nullptr, KTOP, DIM, DIM, DIM, DIM, DIM,
        (long long)KTOP * DIM, (long long)KTOP * DIM, 0, 0);
    k_gemm<<<dim3((KTOP + 15) / 16, DIM / 64, BSZ), 32, 0, stream>>>(
        tkf, bf_Wv, vb, nullptr, nullptr, KTOP, DIM, DIM, DIM, DIM, DIM,
        (long long)KTOP * DIM, (long long)KTOP * DIM, 0, 0);
    k_attn<<<BSZ * NH, 128, 0, stream>>>(qb, kb, vb, ob);
    k_gemm<<<dim3((NDV + 15) / 16, DIM / 64, BSZ), 32, 0, stream>>>(
        ob, bf_Wo, xres, nullptr, l2p, NDV, DIM, DIM, DIM, DIM, DIM,
        (long long)NDV * DIM, (long long)NDV * DIM, (long long)NDV * DIM, 4);
    k_ln<<<BSZ * NDV, 256, 0, stream>>>(xres, ln2g, ln2b, ln2x);
    k_gemm<<<dim3((NDV + 15) / 16, DFF / 64, BSZ), 32, 0, stream>>>(
        ln2x, bf_W1, hid, b1, nullptr, NDV, DFF, DIM, DIM, DIM, DFF,
        (long long)NDV * DIM, (long long)NDV * DFF, 0, 1 | 2);
    k_gemm<<<dim3((NDV + 15) / 16, DIM / 64, BSZ), 32, 0, stream>>>(
        hid, bf_W2, tbias, b2, xres, NDV, DIM, DFF, DFF, DFF, DIM,
        (long long)NDV * DFF, (long long)NDV * DIM, (long long)NDV * DIM, 1 | 4);

    // --- refined text cache ---
    k_refine<<<NCLS * NDV, 256, 0, stream>>>(tc, tbias, labels, rn, bf_rn, o_rn);
    k_rmn<<<NCLS, 256, 0, stream>>>(rn, rmn);
    k_vmat<<<NCLS, 64, 0, stream>>>(rmn, rn, vmat);
    k_base<<<BSZ, 128, 0, stream>>>(wag, rmn, basew);
    k_w<<<BSZ, 256, 0, stream>>>(wag, adapted, wbuf);
    k_wtopk<<<BSZ, 256, 0, stream>>>(adapted, rn /* row (c=0,nd=0) */, wbuf, wtopk);
    k_dense<<<dim3(BSZ, NCLS), 256, 0, stream>>>(adapted, bf_rn, vmat, wtopk, basew, interR);

    // --- cache branch (chunked) ---
    for (int ch = 0; ch < NCACHE / CHUNK; ch++) {
        k_gemm<<<dim3((CHUNK * NP) / 16, DIM / 64, 1), 32, 0, stream>>>(
            rawc + (size_t)ch * CHUNK * NP * DIM, bf_Wa, craw, nullptr, nullptr,
            CHUNK * NP, DIM, DIM, DIM, DIM, DIM, 0, 0, 0, 0);
        k_ckeys<<<CHUNK, 256, 0, stream>>>(craw, btm, clab, ch, ckeys);
    }
    k_intra<<<BSZ, 256, 0, stream>>>(qkbuf, ckeys, cv, o_intra);
    k_final<<<(BSZ * NCLS + 255) / 256, 256, 0, stream>>>(interR, o_intra, lsp, esp,
                                                          o_logits, o_inter);
}
